// IKNet1_31971736551660
// MI455X (gfx1250) — compile-verified, matrix operand order checked by reading, not run on hardware
//
#include <hip/hip_runtime.h>
#include <hip/hip_bf16.h>

// ---------------------------------------------------------------------------
// Sizes
// ---------------------------------------------------------------------------
#define BATCH   8192
#define J       21
#define H       4
#define C       64
#define FEAT    (H * C)          // 256
#define MROWS   (BATCH * J)      // 172032 rows = 10752 tiles of 16 = 5376 pairs
#define NEG_SLOPE 0.2f

typedef __attribute__((ext_vector_type(16))) __bf16 v16bf;
typedef __attribute__((ext_vector_type(8)))  __bf16 v8bf;
typedef __attribute__((ext_vector_type(8)))  float  v8f;

__device__ __forceinline__ __bf16 f2bf(float f) { return (__bf16)f; }
__device__ __forceinline__ float  bf2f(__bf16 h) { return (float)h; }

// Parent of each node in the hand-skeleton tree (-1 for root). Every node's
// incoming-edge set is {self} or {self, parent} -> softmax over <=2 terms.
__device__ __constant__ int PAR[J] = {
    -1, 0, 1, 2, 3, 0, 5, 6, 7, 0, 9, 10, 11, 0, 13, 14, 15, 0, 17, 18, 19};

// ---------------------------------------------------------------------------
// Prep: W2 / W3 (fp32 [K=256, N=256]) -> bf16 transposed [N, K]
// ---------------------------------------------------------------------------
__global__ void __launch_bounds__(256)
prep_weights(const float* __restrict__ W2, const float* __restrict__ W3,
             __bf16* __restrict__ W2t, __bf16* __restrict__ W3t) {
    int idx = blockIdx.x * 256 + threadIdx.x;      // 2 * 65536
    const float* W = (idx < 65536) ? W2 : W3;
    __bf16*      D = (idx < 65536) ? W2t : W3t;
    int i = idx & 65535;
    int n = i >> 8, k = i & 255;
    D[n * 256 + k] = f2bf(W[k * 256 + n]);
}

// ---------------------------------------------------------------------------
// Layer 1: build x0 = [joints(3) | grot(6)] and GEMM with W1 [9,256] (K=9).
// Memory-bound; plain VALU. Output raw h1 in bf16.
// ---------------------------------------------------------------------------
__global__ void __launch_bounds__(256)
l1_gemm(const float* __restrict__ joints, const float* __restrict__ grot,
        const float* __restrict__ W1, __bf16* __restrict__ hout) {
    long long idx = (long long)blockIdx.x * 256 + threadIdx.x; // MROWS*256
    int row = (int)(idx >> 8);
    int col = (int)(idx & 255);
    int b = row / J, n = row % J;
    float x0[9];
    const float* jp = joints + (long long)(b * J + n) * 3;
    x0[0] = jp[0]; x0[1] = jp[1]; x0[2] = jp[2];
    const float* gp = grot + (long long)b * 6;
#pragma unroll
    for (int i = 0; i < 6; ++i) x0[3 + i] = gp[i];
    float s = 0.f;
#pragma unroll
    for (int r = 0; r < 9; ++r) s += x0[r] * W1[r * 256 + col];
    hout[idx] = f2bf(s);
}

// ---------------------------------------------------------------------------
// WMMA bf16 GEMM: C[M,256] = A[M,256] @ Bt^T, Bt stored [N=256, K=256].
// Register-blocked M x 2: one wave -> 32(M) x 64(N) output tile.
// Per k-step (K += 32): 2 A-frags + 4 B-frags -> 8 independent WMMAs; every
// B fragment feeds two v_wmma ops so load latency overlaps matrix issue.
// Block = 8 waves -> 2 row-pairs x full N. Grid = MROWS / 64.
// ---------------------------------------------------------------------------
__global__ void __launch_bounds__(256)
gemm_wmma_bf16(const __bf16* __restrict__ A, const __bf16* __restrict__ Bt,
               __bf16* __restrict__ Cm) {
    const int K = 256, N = 256;
    int wave  = threadIdx.x >> 5;              // 0..7
    int lane  = threadIdx.x & 31;
    int mpair = blockIdx.x * 2 + (wave >> 2);  // pair of 16-row tiles (32 rows)
    int nbase = (wave & 3) * 64;
    int l16   = lane & 15;
    int hi    = lane >> 4;                     // 0 or 1

    // A-fragment addressing per ISA 16-bit A layout:
    //  lanes 0-15 : row = l16, K-halves {0..7, 16..23}
    //  lanes 16-31: row = l16, K-halves {8..15, 24..31}
    const __bf16* ap0 = A + (size_t)(mpair * 32 + l16) * K + hi * 8;
    const __bf16* ap1 = ap0 + 16 * K;
    const __bf16* bp  = Bt + (size_t)(nbase + l16) * K + hi * 8;

    v8f acc0[4] = {}, acc1[4] = {};
#pragma unroll
    for (int kk = 0; kk < K; kk += 32) {
        union { v8bf h[2]; v16bf v; } a0, a1;
        a0.h[0] = *(const v8bf*)(ap0 + kk);
        a0.h[1] = *(const v8bf*)(ap0 + kk + 16);
        a1.h[0] = *(const v8bf*)(ap1 + kk);
        a1.h[1] = *(const v8bf*)(ap1 + kk + 16);
#pragma unroll
        for (int j = 0; j < 4; ++j) {
            union { v8bf h[2]; v16bf v; } b;
            const __bf16* bj = bp + (size_t)(j * 16) * K + kk;
            b.h[0] = *(const v8bf*)(bj);
            b.h[1] = *(const v8bf*)(bj + 16);
            acc0[j] = __builtin_amdgcn_wmma_f32_16x16x32_bf16(
                false, a0.v, false, b.v, (short)0, acc0[j], false, false);
            acc1[j] = __builtin_amdgcn_wmma_f32_16x16x32_bf16(
                false, a1.v, false, b.v, (short)0, acc1[j], false, false);
        }
    }

    // C layout: VGPR r -> row (hi*8 + r), col = nbase + j*16 + l16.
    int row0 = mpair * 32 + hi * 8;
#pragma unroll
    for (int j = 0; j < 4; ++j) {
        int col = nbase + j * 16 + l16;
#pragma unroll
        for (int r = 0; r < 8; ++r) {
            Cm[(size_t)(row0 + r) * N + col]      = f2bf(acc0[j][r]);
            Cm[(size_t)(row0 + 16 + r) * N + col] = f2bf(acc1[j][r]);
        }
    }
}

// ---------------------------------------------------------------------------
// GAT attention epilogue. One block (256 thr) per batch element.
// MODE 0: concat heads, +bias, ReLU -> bf16 [B, 21, 256]
// MODE 1: mean over heads, +b3     -> fp32 [B, 21, 64]
// ---------------------------------------------------------------------------
template <int MODE>
__global__ void __launch_bounds__(256)
gat_attn(const __bf16* __restrict__ h, const float* __restrict__ a_src,
         const float* __restrict__ a_dst, const float* __restrict__ bias,
         __bf16* __restrict__ xout, float* __restrict__ x3out) {
    __shared__ float hs[J * FEAT];            // 21x256 fp32 tile (~21 KB)
    __shared__ float als[J * H], ald[J * H];
    __shared__ float wself[J * H], wpar[J * H];

    int b = blockIdx.x, tid = threadIdx.x;
    const __bf16* hb = h + (size_t)b * (J * FEAT);
    for (int i = tid; i < J * FEAT; i += 256) hs[i] = bf2f(hb[i]);
    __syncthreads();

    // attention logits: 21 nodes x 4 heads x {src,dst} = 168 length-64 dots
    if (tid < J * H * 2) {
        int n = tid >> 3, rem = tid & 7, hd = rem >> 1, ty = rem & 1;
        const float* a = ty ? a_dst : a_src;
        float s = 0.f;
#pragma unroll 8
        for (int c = 0; c < C; ++c) s += hs[n * FEAT + hd * C + c] * a[hd * C + c];
        (ty ? ald : als)[n * H + hd] = s;
    }
    __syncthreads();

    // softmax over {self, parent} incoming edges
    if (tid < J * H) {
        int n = tid >> 2, hd = tid & 3;
        int p = PAR[n];
        float adn = ald[n * H + hd];
        float es = als[n * H + hd] + adn;
        es = es > 0.f ? es : NEG_SLOPE * es;
        if (p < 0) {
            wself[tid] = 1.f; wpar[tid] = 0.f;
        } else {
            float ep = als[p * H + hd] + adn;
            ep = ep > 0.f ? ep : NEG_SLOPE * ep;
            float m  = fmaxf(es, ep);
            float ps = __expf(es - m), pp = __expf(ep - m);
            float inv = 1.f / (ps + pp);
            wself[tid] = ps * inv; wpar[tid] = pp * inv;
        }
    }
    __syncthreads();

    if (MODE == 0) {
        int c = tid, hd = c >> 6;
        float bcol = bias[c];
        for (int n = 0; n < J; ++n) {
            int p = PAR[n];
            float v = wself[n * H + hd] * hs[n * FEAT + c];
            if (p >= 0) v += wpar[n * H + hd] * hs[p * FEAT + c];
            v += bcol;
            v = v > 0.f ? v : 0.f;
            xout[(size_t)b * (J * FEAT) + n * FEAT + c] = f2bf(v);
        }
    } else {
        int c = tid & 63;
        float bcol = bias[c];
        for (int n0 = 0; n0 < J; n0 += 4) {
            int n = n0 + (tid >> 6);
            if (n < J) {
                int p = PAR[n];
                float v = 0.f;
#pragma unroll
                for (int hd = 0; hd < H; ++hd) {
                    float t = wself[n * H + hd] * hs[n * FEAT + hd * C + c];
                    if (p >= 0) t += wpar[n * H + hd] * hs[p * FEAT + hd * C + c];
                    v += t;
                }
                x3out[(size_t)b * (J * C) + n * C + c] = 0.25f * v + bcol;
            }
        }
    }
}

// ---------------------------------------------------------------------------
// Heads: rot_6ds = x3 @ Wt + bt ; pg = relu(mean_n(x3) @ Wg1 + bg1) @ Wg2 + bg2
// One block per batch element.
// ---------------------------------------------------------------------------
__global__ void __launch_bounds__(256)
heads_kernel(const float* __restrict__ x3, const float* __restrict__ Wt,
             const float* __restrict__ bt, const float* __restrict__ Wg1,
             const float* __restrict__ bg1, const float* __restrict__ Wg2,
             const float* __restrict__ bg2, float* __restrict__ out) {
    __shared__ float xl[J * C];
    __shared__ float pooled[C];
    __shared__ float hid[C];
    int b = blockIdx.x, tid = threadIdx.x;
    const float* xb = x3 + (size_t)b * (J * C);
    for (int i = tid; i < J * C; i += 256) xl[i] = xb[i];
    __syncthreads();

    if (tid < J * 6) {                       // rot_6ds
        int n = tid / 6, o = tid % 6;
        float s = bt[o];
#pragma unroll 8
        for (int c = 0; c < C; ++c) s += xl[n * C + c] * Wt[c * 6 + o];
        out[(size_t)b * (J * 6) + tid] = s;
    }
    if (tid < C) {                           // mean pool over nodes
        float s = 0.f;
        for (int n = 0; n < J; ++n) s += xl[n * C + tid];
        pooled[tid] = s * (1.f / (float)J);
    }
    __syncthreads();
    if (tid < C) {
        float s = bg1[tid];
#pragma unroll 8
        for (int c = 0; c < C; ++c) s += pooled[c] * Wg1[c * C + tid];
        hid[tid] = s > 0.f ? s : 0.f;
    }
    __syncthreads();
    if (tid < 6) {
        float s = bg2[tid];
#pragma unroll 8
        for (int c = 0; c < C; ++c) s += hid[c] * Wg2[c * 6 + tid];
        out[(size_t)BATCH * (J * 6) + (size_t)b * 6 + tid] = s;
    }
}

// ---------------------------------------------------------------------------
// Launch
// ---------------------------------------------------------------------------
extern "C" void kernel_launch(void* const* d_in, const int* in_sizes, int n_in,
                              void* d_out, int out_size, void* d_ws, size_t ws_size,
                              hipStream_t stream) {
    const float* joints = (const float*)d_in[0];
    const float* grot   = (const float*)d_in[1];
    const float* W1     = (const float*)d_in[2];
    const float* a_s1   = (const float*)d_in[3];
    const float* a_d1   = (const float*)d_in[4];
    const float* b1     = (const float*)d_in[5];
    const float* W2     = (const float*)d_in[6];
    const float* a_s2   = (const float*)d_in[7];
    const float* a_d2   = (const float*)d_in[8];
    const float* b2     = (const float*)d_in[9];
    const float* W3     = (const float*)d_in[10];
    const float* a_s3   = (const float*)d_in[11];
    const float* a_d3   = (const float*)d_in[12];
    const float* b3     = (const float*)d_in[13];
    const float* Wt     = (const float*)d_in[14];
    const float* bt     = (const float*)d_in[15];
    const float* Wg1    = (const float*)d_in[16];
    const float* bg1    = (const float*)d_in[17];
    const float* Wg2    = (const float*)d_in[18];
    const float* bg2    = (const float*)d_in[19];
    float* out = (float*)d_out;

    // Workspace layout (16B-aligned sections)
    const size_t HBF_BYTES = (size_t)MROWS * FEAT * sizeof(__bf16); // 88 MB
    const size_t X3_BYTES  = (size_t)BATCH * J * C * sizeof(float); // 44 MB
    char* ws = (char*)d_ws;
    __bf16* h_bf = (__bf16*)(ws);
    __bf16* x_bf = (__bf16*)(ws + HBF_BYTES);
    float*  x3   = (float*) (ws + 2 * HBF_BYTES);
    __bf16* W2t  = (__bf16*)(ws + 2 * HBF_BYTES + X3_BYTES);
    __bf16* W3t  = W2t + 256 * 256;

    // 0) weights -> transposed bf16
    prep_weights<<<512, 256, 0, stream>>>(W2, W3, W2t, W3t);
    // 1) layer-1 GEMM (K=9), raw h1
    l1_gemm<<<MROWS, 256, 0, stream>>>(joints, grot, W1, h_bf);
    // 2) layer-1 attention (+b1, ReLU) -> x_bf
    gat_attn<0><<<BATCH, 256, 0, stream>>>(h_bf, a_s1, a_d1, b1, x_bf, nullptr);
    // 3) layer-2 GEMM (WMMA bf16) -> h_bf
    gemm_wmma_bf16<<<MROWS / 64, 256, 0, stream>>>(x_bf, W2t, h_bf);
    // 4) layer-2 attention (+b2, ReLU) -> x_bf
    gat_attn<0><<<BATCH, 256, 0, stream>>>(h_bf, a_s2, a_d2, b2, x_bf, nullptr);
    // 5) layer-3 GEMM (WMMA bf16) -> h_bf
    gemm_wmma_bf16<<<MROWS / 64, 256, 0, stream>>>(x_bf, W3t, h_bf);
    // 6) layer-3 attention, head-mean (+b3) -> x3 (fp32)
    gat_attn<1><<<BATCH, 256, 0, stream>>>(h_bf, a_s3, a_d3, b3, nullptr, x3);
    // 7) output heads
    heads_kernel<<<BATCH, 256, 0, stream>>>(x3, Wt, bt, Wg1, bg1, Wg2, bg2, out);

    (void)in_sizes; (void)n_in; (void)out_size; (void)ws_size;
}
